// GPT2AttentionMixer_20744692039800
// MI455X (gfx1250) — matrix operand order repaired and checked by reading, not compiled, not run on hardware
//
#include <hip/hip_runtime.h>

// ---------------------------------------------------------------------------
// GPT-2 attention mixer for gfx1250 (CDNA5): bf16 WMMA + double-buffered
// async-to-LDS DMA staging. B=2, S=2048, E=1024, H=16, HD=64.
// ---------------------------------------------------------------------------

typedef __bf16 v16bf __attribute__((ext_vector_type(16)));
typedef float  v8f   __attribute__((ext_vector_type(8)));
typedef int    i32x4 __attribute__((vector_size(4 * sizeof(int))));

union ABOp { unsigned int u[8]; v16bf v; };

#if defined(__gfx1250__) && defined(__has_builtin)
#  if __has_builtin(__builtin_amdgcn_global_load_async_to_lds_b128) && \
      __has_builtin(__builtin_amdgcn_s_wait_asynccnt)
#    define HAVE_ASYNC 1
#  endif
#endif
#ifndef HAVE_ASYNC
#  define HAVE_ASYNC 0
#endif

#define AS1 __attribute__((address_space(1)))
#define AS3 __attribute__((address_space(3)))

// Copy 16 bytes global -> LDS via the CDNA5 async-to-LDS DMA (ASYNCcnt).
__device__ __forceinline__ void copy16_g2l(const void* g, void* l) {
#if HAVE_ASYNC
  __builtin_amdgcn_global_load_async_to_lds_b128(
      (AS1 i32x4*)(unsigned long long)g,
      (AS3 i32x4*)(unsigned int)(unsigned long long)l, 0, 0);
#else
  *(int4*)l = *(const int4*)g;
#endif
}

__device__ __forceinline__ void async_wait_all() {
#if HAVE_ASYNC
  __builtin_amdgcn_s_wait_asynccnt(0);
#endif
}
// Async ops complete in order: waiting "<= N" retires everything older than
// the newest N in-flight ops. Used for double-buffered pipelining.
__device__ __forceinline__ void async_wait_keep3() {
#if HAVE_ASYNC
  __builtin_amdgcn_s_wait_asynccnt(3);
#endif
}
__device__ __forceinline__ void async_wait_keep16() {
#if HAVE_ASYNC
  __builtin_amdgcn_s_wait_asynccnt(16);
#endif
}

__device__ __forceinline__ unsigned short f32_to_bf16(float f) {
  unsigned int u = __float_as_uint(f);
  u += 0x7FFFu + ((u >> 16) & 1u);   // round-to-nearest-even
  return (unsigned short)(u >> 16);
}

// ---------------------------------------------------------------------------
// f32 -> bf16 conversion (grid-stride)
// ---------------------------------------------------------------------------
__global__ void __launch_bounds__(256)
cvt_bf16(const float* __restrict__ in, unsigned short* __restrict__ out, int n) {
  int i = blockIdx.x * blockDim.x + threadIdx.x;
  int stride = gridDim.x * blockDim.x;
  for (; i < n; i += stride) out[i] = f32_to_bf16(in[i]);
}

// f32 [K][N] -> bf16 transposed [N][K] (one-time weight prep so GEMM B-tiles
// become contiguous rows for async b128 staging)
__global__ void __launch_bounds__(256)
cvt_bf16_t(const float* __restrict__ in, unsigned short* __restrict__ out,
           int K, int N) {
  int n = blockIdx.x * 64 + threadIdx.x;   // block (64,4)
  int k = blockIdx.y * 4 + threadIdx.y;
  if (n < N && k < K) out[(size_t)n * K + k] = f32_to_bf16(in[(size_t)k * N + n]);
}

// ---------------------------------------------------------------------------
// bf16 GEMM: C[M,N] = A[M,K] * BT[N,K]^T + bias[N]   (B given pre-transposed)
// Block: 256 threads (8 waves). Block tile 128x64, wave tile 16x64, K-step 32.
// Double-buffered LDS: next tile's DMA overlaps current tile's WMMAs.
// mode 0: QKV epilogue -> scatter Q (x0.125, [B,H,S,64]), K ([B,H,S,64]),
//                         V transposed ([B,H,64,S]) as bf16.
// mode 1: plain f32 row-major output.
// ---------------------------------------------------------------------------
__global__ void __launch_bounds__(256)
gemm_bf16(const unsigned short* __restrict__ A,
          const unsigned short* __restrict__ BT,
          const float* __restrict__ bias,
          int N, int K, int mode,
          unsigned short* __restrict__ qb,
          unsigned short* __restrict__ kb,
          unsigned short* __restrict__ vt,
          float* __restrict__ outf) {
  __shared__ unsigned short lA[2][128][32];  // 2 x (128 rows x 32 bf16)
  __shared__ unsigned short lB[2][64][32];   // 2 x (64 cols  x 32 bf16)

  const int tid  = threadIdx.x;
  const int lane = tid & 31;
  const int w    = tid >> 5;
  const int m0   = blockIdx.y * 128;
  const int n0   = blockIdx.x * 64;

  const int r    = lane & 15;
  const int hi8  = (lane >> 4) << 3;   // A-operand K offset for upper half-wave
  const int kb16 = (lane >> 4) << 4;   // unused: B operand also uses hi8 as its K offset

  // Stage one 128x32 A tile + 64x32 B tile: exactly 3 async ops per lane/wave.
  auto stage = [&](int k0, int buf) {
    for (int c = tid; c < 128 * 4; c += 256) {
      int rr = c >> 2, seg = c & 3;
      copy16_g2l(A + (size_t)(m0 + rr) * K + k0 + seg * 8, &lA[buf][rr][seg * 8]);
    }
    for (int c = tid; c < 64 * 4; c += 256) {
      int nn = c >> 2, seg = c & 3;
      copy16_g2l(BT + (size_t)(n0 + nn) * K + k0 + seg * 8, &lB[buf][nn][seg * 8]);
    }
  };

  const v8f vzero = {0, 0, 0, 0, 0, 0, 0, 0};
  v8f acc[4];
#pragma unroll
  for (int t = 0; t < 4; ++t) acc[t] = vzero;

  stage(0, 0);
  int buf = 0;
  for (int k0 = 0; k0 < K; k0 += 32) {
    __syncthreads();  // all waves done reading buf^1 (previous compute)
    if (k0 + 32 < K) {
      stage(k0 + 32, buf ^ 1);   // overlap next tile's DMA with this compute
      async_wait_keep3();        // retires stage(k0); keeps stage(k0+32) flying
    } else {
      async_wait_all();
    }
    __syncthreads();  // stage(k0) visible to all waves

    // A operand: lane holds row M=lane&15; K interleave 0-7/16-23 | 8-15/24-31
    ABOp a;
#pragma unroll
    for (int p = 0; p < 8; ++p) {
      int kk = 2 * p + ((p >= 4) ? 8 : 0) + hi8;
      a.u[p] = *(const unsigned int*)&lA[buf][w * 16 + r][kk];
    }
#pragma unroll
    for (int t = 0; t < 4; ++t) {
      ABOp bm;
#pragma unroll
      for (int p = 0; p < 8; ++p) {
        int kk = 2 * p + ((p >= 4) ? 8 : 0) + hi8;  // B: K interleave 0-7/16-23 | 8-15/24-31
        bm.u[p] = *(const unsigned int*)&lB[buf][t * 16 + r][kk];
      }
      acc[t] = __builtin_amdgcn_wmma_f32_16x16x32_bf16(
          false, a.v, false, bm.v, (short)0, acc[t], false, false);
    }
    buf ^= 1;
  }

  // epilogue: C layout VGPR rr -> row rr + 8*(lane>=16), col = lane&15
#pragma unroll
  for (int t = 0; t < 4; ++t) {
#pragma unroll
    for (int rr = 0; rr < 8; ++rr) {
      int mrow = m0 + w * 16 + rr + hi8;
      int ncol = n0 + t * 16 + (lane & 15);
      float val = acc[t][rr] + bias[ncol];
      if (mode == 0) {
        int bb = mrow >> 11, ss = mrow & 2047;
        int which = ncol >> 10, e = ncol & 1023;
        int hh = e >> 6, dd = e & 63;
        size_t headrow = ((size_t)((bb << 4) + hh) * 2048 + ss) * 64 + dd;
        if (which == 0)      qb[headrow] = f32_to_bf16(val * 0.125f);  // 1/sqrt(64)
        else if (which == 1) kb[headrow] = f32_to_bf16(val);
        else                 vt[((size_t)((bb << 4) + hh) * 64 + dd) * 2048 + ss] =
                                 f32_to_bf16(val);
      } else {
        outf[(size_t)mrow * N + ncol] = val;
      }
    }
  }
}

// ---------------------------------------------------------------------------
// Flash attention: 128 threads (4 waves), each wave owns 16 query rows.
// Q pre-scaled; K in [B,H,S,64]; V transposed [B,H,64,S]. 32-key tiles,
// double-buffered per-wave async DMA (16 ops/stage, ASYNCcnt in-order).
// ---------------------------------------------------------------------------
__global__ void __launch_bounds__(128)
attn_kernel(const unsigned short* __restrict__ qbm,
            const unsigned short* __restrict__ kbm,
            const unsigned short* __restrict__ vtm,
            unsigned short* __restrict__ ao) {
  __shared__ unsigned short ldsK[4][2][32][64];  // per-wave K tiles (rows=key)
  __shared__ unsigned short ldsV[4][2][64][32];  // per-wave V tiles (rows=d)
  __shared__ unsigned short ldsP[4][16][32];     // per-wave P staging

  const int tid  = threadIdx.x;
  const int lane = tid & 31;
  const int w    = tid >> 5;
  const int q0   = blockIdx.x * 64;
  const int bh   = blockIdx.y;
  const int b    = bh >> 4, h = bh & 15;

  const size_t qkbase = (size_t)bh * 2048 * 64;
  const size_t vbase  = (size_t)bh * 64 * 2048;

  const int r    = lane & 15;
  const int hi8  = (lane >> 4) << 3;
  const int kb16 = (lane >> 4) << 4;
  const int qr0  = q0 + w * 16;

  const unsigned int* qg = (const unsigned int*)qbm;

  // Stage K tile (32 x 128B) + V tile (64 x 64B): 16 async ops per wave.
  auto stageKV = [&](int key0, int buf) {
#pragma unroll
    for (int it = 0; it < 8; ++it) {
      int c = it * 32 + lane;
      int row = c >> 3, seg = c & 7;
      copy16_g2l(kbm + qkbase + (size_t)(key0 + row) * 64 + seg * 8,
                 &ldsK[w][buf][row][seg * 8]);
    }
#pragma unroll
    for (int it = 0; it < 8; ++it) {
      int c = it * 32 + lane;
      int row = c >> 2, seg = c & 3;
      copy16_g2l(vtm + vbase + (size_t)row * 2048 + key0 + seg * 8,
                 &ldsV[w][buf][row][seg * 8]);
    }
  };

  // Q A-operands: two K-dim-32 chunks covering HD=64, held for the whole loop
  ABOp qa[2];
#pragma unroll
  for (int c = 0; c < 2; ++c) {
#pragma unroll
    for (int p = 0; p < 8; ++p) {
      int kk = 2 * p + ((p >= 4) ? 8 : 0) + hi8 + c * 32;
      qa[c].u[p] = qg[(qkbase + (size_t)(qr0 + r) * 64 + kk) >> 1];
    }
  }

  const v8f vzero = {0, 0, 0, 0, 0, 0, 0, 0};
  v8f o[4];
#pragma unroll
  for (int t = 0; t < 4; ++t) o[t] = vzero;
  float mrow[8], lrow[8];
#pragma unroll
  for (int i = 0; i < 8; ++i) { mrow[i] = -3.0e38f; lrow[i] = 0.0f; }

  const int nkt = (qr0 + 16 + 31) >> 5;  // causal: keys up to qr0+15
  stageKV(0, 0);
  int buf = 0;
  for (int kt = 0; kt < nkt; ++kt) {
    const int key0 = kt * 32;
    // issue next tile's DMA before waiting; wave-private so no barriers
    if (kt + 1 < nkt) {
      stageKV((kt + 1) * 32, buf ^ 1);
      async_wait_keep16();   // retires stage(kt); keeps stage(kt+1) in flight
    } else {
      async_wait_all();
    }

    // scores = Q K^T for two 16-key sub-tiles
    float sc[2][8];
#pragma unroll
    for (int t2 = 0; t2 < 2; ++t2) {
      ABOp kbop[2];
#pragma unroll
      for (int c = 0; c < 2; ++c) {
#pragma unroll
        for (int p = 0; p < 8; ++p) {
          int kk = 2 * p + ((p >= 4) ? 8 : 0) + hi8;  // d offset within chunk
          kbop[c].u[p] =
              *(const unsigned int*)&ldsK[w][buf][t2 * 16 + r][c * 32 + kk];
        }
      }
      v8f s = vzero;
      s = __builtin_amdgcn_wmma_f32_16x16x32_bf16(false, qa[0].v, false, kbop[0].v,
                                                  (short)0, s, false, false);
      s = __builtin_amdgcn_wmma_f32_16x16x32_bf16(false, qa[1].v, false, kbop[1].v,
                                                  (short)0, s, false, false);
#pragma unroll
      for (int rr = 0; rr < 8; ++rr) {
        int q = qr0 + rr + hi8;
        int key = key0 + t2 * 16 + r;
        sc[t2][rr] = (key <= q) ? s[rr] : -1.0e30f;  // causal mask
      }
    }

    // online softmax update (per row; stats replicated across 16-lane halves)
#pragma unroll
    for (int rr = 0; rr < 8; ++rr) {
      float tm = fmaxf(sc[0][rr], sc[1][rr]);
#pragma unroll
      for (int off = 8; off >= 1; off >>= 1)
        tm = fmaxf(tm, __shfl_xor(tm, off, 32));
      float mnew  = fmaxf(mrow[rr], tm);
      float scale = __expf(mrow[rr] - mnew);
      mrow[rr] = mnew;
      float p0 = __expf(sc[0][rr] - mnew);
      float p1 = __expf(sc[1][rr] - mnew);
      float rs = p0 + p1;
#pragma unroll
      for (int off = 8; off >= 1; off >>= 1)
        rs += __shfl_xor(rs, off, 32);
      lrow[rr] = lrow[rr] * scale + rs;
#pragma unroll
      for (int t = 0; t < 4; ++t) o[t][rr] *= scale;
      int prow = rr + hi8;
      ldsP[w][prow][(lane & 15)]      = f32_to_bf16(p0);
      ldsP[w][prow][16 + (lane & 15)] = f32_to_bf16(p1);
    }

    // P (16x32 bf16) re-layout via LDS into A operand
    ABOp pa;
#pragma unroll
    for (int p = 0; p < 8; ++p) {
      int kk = 2 * p + ((p >= 4) ? 8 : 0) + hi8;
      pa.u[p] = *(const unsigned int*)&ldsP[w][lane & 15][kk];
    }
    // O += P * V  (V tile rows = d, key pairs contiguous)
#pragma unroll
    for (int t = 0; t < 4; ++t) {
      ABOp vbop;
#pragma unroll
      for (int p = 0; p < 8; ++p) {
        int kk = 2 * p + ((p >= 4) ? 8 : 0) + hi8;  // key offset within tile
        vbop.u[p] = *(const unsigned int*)&ldsV[w][buf][t * 16 + r][kk];
      }
      o[t] = __builtin_amdgcn_wmma_f32_16x16x32_bf16(false, pa.v, false, vbop.v,
                                                     (short)0, o[t], false, false);
    }
    buf ^= 1;
  }

  // normalize and store merged-head output [B,S,E] bf16
#pragma unroll
  for (int t = 0; t < 4; ++t) {
#pragma unroll
    for (int rr = 0; rr < 8; ++rr) {
      int srow = qr0 + rr + hi8;
      int col  = h * 64 + t * 16 + (lane & 15);
      ao[((size_t)b * 2048 + srow) * 1024 + col] = f32_to_bf16(o[t][rr] / lrow[rr]);
    }
  }
}

// ---------------------------------------------------------------------------
// Launch
// ---------------------------------------------------------------------------
extern "C" void kernel_launch(void* const* d_in, const int* in_sizes, int n_in,
                              void* d_out, int out_size, void* d_ws, size_t ws_size,
                              hipStream_t stream) {
  const float* x      = (const float*)d_in[0];
  const float* w_qkv  = (const float*)d_in[1];
  const float* b_qkv  = (const float*)d_in[2];
  const float* w_proj = (const float*)d_in[3];
  const float* b_proj = (const float*)d_in[4];
  float* out = (float*)d_out;

  unsigned short* ws = (unsigned short*)d_ws;
  // workspace layout (u16 elements):
  unsigned short* xb      = ws;             // 4096*1024      [0,       4194304)
  unsigned short* wqkvT   = ws + 4194304;   // [3072][1024]   [4194304, 7340032)
  unsigned short* wprojT  = ws + 7340032;   // [1024][1024]   [7340032, 8388608)
  unsigned short* qb      = ws + 8388608;   // [B,H,S,64]
  unsigned short* kb      = ws + 12582912;  // [B,H,S,64]
  unsigned short* vt      = ws + 16777216;  // [B,H,64,S]
  unsigned short* ao      = ws;             // aliases xb (dead after QKV gemm)

  cvt_bf16<<<1024, 256, 0, stream>>>(x, xb, 2 * 2048 * 1024);
  cvt_bf16_t<<<dim3(48, 256), dim3(64, 4), 0, stream>>>(w_qkv,  wqkvT,  1024, 3072);
  cvt_bf16_t<<<dim3(16, 256), dim3(64, 4), 0, stream>>>(w_proj, wprojT, 1024, 1024);

  // QKV projection: M=4096, N=3072, K=1024
  gemm_bf16<<<dim3(48, 32), 256, 0, stream>>>(xb, wqkvT, b_qkv, 3072, 1024, 0,
                                              qb, kb, vt, nullptr);
  // flash attention: 32 q-blocks x 32 (b,h)
  attn_kernel<<<dim3(32, 32), 128, 0, stream>>>(qb, kb, vt, ao);
  // output projection: M=4096, N=1024, K=1024
  gemm_bf16<<<dim3(16, 32), 256, 0, stream>>>(ao, wprojT, b_proj, 1024, 1024, 1,
                                              nullptr, nullptr, nullptr, out);
}
